// InfoStill_84516366451098
// MI455X (gfx1250) — matrix operand order repaired and checked by reading, not compiled, not run on hardware
//
#include <hip/hip_runtime.h>

// ---------------- problem constants ----------------
constexpr int kB = 2;
constexpr int kS = 2048;
constexpr int kE = 1024;
constexpr int kH = 16;
constexpr int kD = 64;     // kE / kH

typedef __bf16 bf16;
typedef __attribute__((ext_vector_type(16))) __bf16 v16bf;
typedef __attribute__((ext_vector_type(8)))  float  v8f;
typedef __attribute__((ext_vector_type(4)))  int    v4i;

union FragAB { uint4 q[2]; v16bf v; };   // 16 bf16 = 32B  (A or B operand)
union FragC  { v8f v; float f[8]; };     // 16x16 f32 C/D tile slice per lane

// ---- CDNA5 async memory->LDS staging (GLOBAL_LOAD_ASYNC_TO_LDS_B128) ----
#if defined(__gfx1250__) && __has_builtin(__builtin_amdgcn_global_load_async_to_lds_b128)
#define HAVE_ASYNC_COPY 1
typedef __attribute__((address_space(1))) v4i gv4i;   // global int4
typedef __attribute__((address_space(3))) v4i lv4i;   // LDS int4
__device__ __forceinline__ void async_cp16(const bf16* g, bf16* l) {
  __builtin_amdgcn_global_load_async_to_lds_b128((gv4i*)(bf16*)g, (lv4i*)l, 0, 0);
}
__device__ __forceinline__ void async_wait0() {
#if __has_builtin(__builtin_amdgcn_s_wait_asynccnt)
  __builtin_amdgcn_s_wait_asynccnt(0);
#else
  asm volatile("s_wait_asynccnt 0x0" ::: "memory");
#endif
}
#endif

// =====================================================================
// Kernel 1: LayerNorm (f32 in) -> bf16 activations h[4096][1024]
// =====================================================================
__global__ __launch_bounds__(256) void ln_kernel(const float* __restrict__ x,
                                                 const float* __restrict__ gamma,
                                                 const float* __restrict__ beta,
                                                 bf16* __restrict__ h) {
  __shared__ float red[256];
  const int row = blockIdx.x;
  const int tid = threadIdx.x;
  const float* xr = x + (size_t)row * kE;

  float v[4];
  float s = 0.f, ss = 0.f;
  #pragma unroll
  for (int i = 0; i < 4; ++i) {
    const int j = tid + i * 256;
    v[i] = xr[j];
    s  += v[i];
    ss += v[i] * v[i];
  }
  red[tid] = s; __syncthreads();
  for (int o = 128; o > 0; o >>= 1) { if (tid < o) red[tid] += red[tid + o]; __syncthreads(); }
  const float mean = red[0] * (1.f / kE);
  __syncthreads();
  red[tid] = ss; __syncthreads();
  for (int o = 128; o > 0; o >>= 1) { if (tid < o) red[tid] += red[tid + o]; __syncthreads(); }
  const float var = red[0] * (1.f / kE) - mean * mean;
  const float inv = rsqrtf(var + 1e-5f);

  bf16* hr = h + (size_t)row * kE;
  #pragma unroll
  for (int i = 0; i < 4; ++i) {
    const int j = tid + i * 256;
    hr[j] = (bf16)((v[i] - mean) * inv * gamma[j] + beta[j]);
  }
}

// =====================================================================
// Kernel 2: transpose f32 [K][N] -> bf16 [N][K]   (weights -> Bt layout)
// =====================================================================
__global__ __launch_bounds__(256) void transpose_cvt_kernel(const float* __restrict__ in,
                                                            bf16* __restrict__ out,
                                                            int K, int N) {
  __shared__ float tile[32][33];
  const int n0 = blockIdx.x * 32;
  const int k0 = blockIdx.y * 32;
  const int tx = threadIdx.x;
  for (int i = threadIdx.y; i < 32; i += 8)
    tile[i][tx] = in[(size_t)(k0 + i) * N + n0 + tx];
  __syncthreads();
  for (int i = threadIdx.y; i < 32; i += 8)
    out[(size_t)(n0 + i) * K + k0 + tx] = (bf16)tile[tx][i];
}

// =====================================================================
// Kernel 3: bf16 WMMA GEMM  C[M][N] = A[M][K] @ Bt[N][K]^T   (f32 out)
// block = 256 threads (8 waves, 2x4), block tile 64(M) x 256(N),
// wave tile 32x64 -> 8 WMMA per 32-K step, K-step 32.
// LDS tiles staged with GLOBAL_LOAD_ASYNC_TO_LDS_B128 when available.
// =====================================================================
__global__ __launch_bounds__(256) void gemm_bf16_kernel(const bf16* __restrict__ A,
                                                        const bf16* __restrict__ Bt,
                                                        float* __restrict__ C,
                                                        int M, int N, int K) {
  __shared__ bf16 lA[64][32];    //  4 KB
  __shared__ bf16 lB[256][32];   // 16 KB
  const int tid  = threadIdx.x;
  const int lane = tid & 31, wave = tid >> 5;
  const int wr = wave >> 2, wc = wave & 3;           // 2 x 4 wave grid
  const int l16 = lane & 15, half = lane >> 4;
  const int lo   = half ? 8  : 0;                    // A-frag K sub-offset
  const int koff = half ? 16 : 0;                    // unused (B-frag uses lo too)
  const int m0 = blockIdx.y * 64;
  const int n0 = blockIdx.x * 256;

  FragC acc[2][4];
  #pragma unroll
  for (int mi = 0; mi < 2; ++mi)
    #pragma unroll
    for (int ni = 0; ni < 4; ++ni)
      #pragma unroll
      for (int r = 0; r < 8; ++r) acc[mi][ni].f[r] = 0.f;

  const int ar = tid >> 2, ac = (tid & 3) * 8;       // staging coords

  for (int kk = 0; kk < K; kk += 32) {
#if defined(HAVE_ASYNC_COPY)
    // ---- direct memory->LDS DMA (ASYNCcnt) ----
    async_cp16(&A[(size_t)(m0 + ar) * K + kk + ac], &lA[ar][ac]);
    #pragma unroll
    for (int i = 0; i < 4; ++i) {
      const int idx = tid + i * 256;
      const int br = idx >> 2, bc = (idx & 3) * 8;
      async_cp16(&Bt[(size_t)(n0 + br) * K + kk + bc], &lB[br][bc]);
    }
    async_wait0();
#else
    // ---- fallback: VGPR round-trip staging ----
    *(uint4*)&lA[ar][ac] = *(const uint4*)&A[(size_t)(m0 + ar) * K + kk + ac];
    #pragma unroll
    for (int i = 0; i < 4; ++i) {
      const int idx = tid + i * 256;
      const int br = idx >> 2, bc = (idx & 3) * 8;
      *(uint4*)&lB[br][bc] = *(const uint4*)&Bt[(size_t)(n0 + br) * K + kk + bc];
    }
#endif
    if (kk + 32 < K) {
      __builtin_prefetch(&A [(size_t)(m0 + ar) * K + kk + 32 + ac], 0, 0);
      __builtin_prefetch(&Bt[(size_t)(n0 + ar) * K + kk + 32 + ac], 0, 0);
    }
    __syncthreads();

    FragAB a[2], b[4];
    #pragma unroll
    for (int mi = 0; mi < 2; ++mi) {
      const bf16* ap = &lA[wr * 32 + mi * 16 + l16][0];
      a[mi].q[0] = *(const uint4*)(ap + lo);
      a[mi].q[1] = *(const uint4*)(ap + lo + 16);
    }
    #pragma unroll
    for (int ni = 0; ni < 4; ++ni) {
      const bf16* bp = &lB[wc * 64 + ni * 16 + l16][0];
      b[ni].q[0] = *(const uint4*)(bp + lo);
      b[ni].q[1] = *(const uint4*)(bp + lo + 16);
    }
    #pragma unroll
    for (int mi = 0; mi < 2; ++mi)
      #pragma unroll
      for (int ni = 0; ni < 4; ++ni)
        acc[mi][ni].v = __builtin_amdgcn_wmma_f32_16x16x32_bf16(
            false, a[mi].v, false, b[ni].v, (short)0, acc[mi][ni].v, false, false);
    __syncthreads();
  }

  #pragma unroll
  for (int mi = 0; mi < 2; ++mi)
    #pragma unroll
    for (int ni = 0; ni < 4; ++ni)
      #pragma unroll
      for (int r = 0; r < 8; ++r) {
        const int row = m0 + wr * 32 + mi * 16 + r + 8 * half;
        const int col = n0 + wc * 64 + ni * 16 + l16;
        C[(size_t)row * N + col] = acc[mi][ni].f[r];
      }
}

// =====================================================================
// Kernel 4: RoPE (position-independent, theta = S * base^-(2i/E)) +
// head split + per-head L2 norm.  Emits:
//   qn[B][H][S][D] bf16   (A operand of Q@K^T)
//   kn[B][H][S][D] bf16   (Bt operand of Q@K^T : N=key row, K=d)
//   vT[B][H][D][S] bf16   (Bt operand of P@V   : N=d row,  K=key)
// =====================================================================
__global__ __launch_bounds__(256) void rope_split_kernel(const float* __restrict__ qkv,
                                                         bf16* __restrict__ qn,
                                                         bf16* __restrict__ kn,
                                                         bf16* __restrict__ vT) {
  const int row = blockIdx.x;            // b*S + s
  const int b = row / kS, s = row % kS;
  const float* qr = qkv + (size_t)row * (3 * kE);
  const float* kr = qr + kE;
  const float* vr = qr + 2 * kE;
  const int tid = threadIdx.x;
  const int hd  = tid >> 4;              // head id: 4 elems/thread stay in-head
  const int j0  = tid * 4;

  float qv[4], kv[4], vv[4];
  float sq = 0.f, sk = 0.f;
  #pragma unroll
  for (int i = 0; i < 4; ++i) {
    const int j = j0 + i;
    const float e2 = (float)(j & ~1);
    // base^-(e2/E) = exp(-ln(10000)*e2/1024)
    const float fr = __expf(e2 * (-9.210340371976184f / 1024.f));
    const float th = 2048.f * fr;
    const float sn = __sinf(th), cs = __cosf(th);
    const float qrot = (j < 512) ? -qr[2 * j + 1] : qr[2 * (j - 512)];
    const float krot = (j < 512) ? -kr[2 * j + 1] : kr[2 * (j - 512)];
    qv[i] = qr[j] * cs + qrot * sn;
    kv[i] = kr[j] * cs + krot * sn;
    vv[i] = vr[j];
    sq += qv[i] * qv[i];
    sk += kv[i] * kv[i];
  }
  // reduce sum-of-squares over the 16 lanes of this head (stays in half-wave)
  #pragma unroll
  for (int m = 8; m >= 1; m >>= 1) {
    sq += __shfl_xor(sq, m);
    sk += __shfl_xor(sk, m);
  }
  const float qi = 1.f / fmaxf(sqrtf(sq), 1e-12f);
  const float ki = 1.f / fmaxf(sqrtf(sk), 1e-12f);

  const size_t bh   = (size_t)(b * kH + hd);
  const size_t baseQK = (bh * kS + s) * kD;
  #pragma unroll
  for (int i = 0; i < 4; ++i) {
    const int d = (j0 + i) & (kD - 1);
    qn[baseQK + d] = (bf16)(qv[i] * qi);
    kn[baseQK + d] = (bf16)(kv[i] * ki);
    vT[(bh * kD + d) * kS + s] = (bf16)vv[i];
  }
}

// =====================================================================
// Kernel 5: causal flash attention, 1 wave per 16-row Q tile.
// grid = (S/128, B*H), 8 independent waves/block.
// Per 32-key step: 4 WMMA (scores, K=64) + online softmax +
// P transpose via per-wave LDS + 4 WMMA (P@V, D=64).
// Output: aout[b*S + q][h*64 + d] bf16 (A operand of final GEMM).
// =====================================================================
__global__ __launch_bounds__(256) void attn_kernel(const bf16* __restrict__ qn,
                                                   const bf16* __restrict__ kn,
                                                   const bf16* __restrict__ vT,
                                                   const float* __restrict__ qkscale,
                                                   bf16* __restrict__ aout) {
  __shared__ bf16 pbuf[8][16][32];   // per-wave P transpose scratch (8 KB)
  const int lane = threadIdx.x & 31;
  const int wave = threadIdx.x >> 5;
  const int bh = blockIdx.y;                    // b*H + h
  const int q0 = blockIdx.x * 128 + wave * 16;  // this wave's 16 query rows
  const int b = bh >> 4, h = bh & 15;
  const float scale = qkscale[0];

  const bf16* qp = qn + (size_t)bh * kS * kD;
  const bf16* kp = kn + (size_t)bh * kS * kD;
  const bf16* vp = vT + (size_t)bh * kD * kS;

  const int l16 = lane & 15, half = lane >> 4;
  const int lo   = half ? 8  : 0;
  const int koff = half ? 16 : 0;

  // Q A-fragments for the two K=32 halves of D=64 (held in registers)
  FragAB aq[2];
  #pragma unroll
  for (int kd = 0; kd < 2; ++kd) {
    const bf16* rp = qp + (size_t)(q0 + l16) * kD + kd * 32;
    aq[kd].q[0] = *(const uint4*)(rp + lo);
    aq[kd].q[1] = *(const uint4*)(rp + lo + 16);
  }

  float rmax[8], rsum[8];
  FragC oacc[4];
  #pragma unroll
  for (int r = 0; r < 8; ++r) { rmax[r] = -1e30f; rsum[r] = 0.f; }
  #pragma unroll
  for (int d4 = 0; d4 < 4; ++d4)
    #pragma unroll
    for (int r = 0; r < 8; ++r) oacc[d4].f[r] = 0.f;

  for (int kb = 0; kb <= q0 + 15; kb += 32) {   // wave-uniform causal bound
    // ---- scores: S[16x32] = Q[16x64] @ K^T[64x32] ----
    FragC sc[2];
    #pragma unroll
    for (int ni = 0; ni < 2; ++ni) {
      #pragma unroll
      for (int r = 0; r < 8; ++r) sc[ni].f[r] = 0.f;
      const bf16* krow = kp + (size_t)(kb + ni * 16 + l16) * kD;
      #pragma unroll
      for (int kd = 0; kd < 2; ++kd) {
        FragAB bk;
        bk.q[0] = *(const uint4*)(krow + kd * 32 + lo);
        bk.q[1] = *(const uint4*)(krow + kd * 32 + lo + 16);
        sc[ni].v = __builtin_amdgcn_wmma_f32_16x16x32_bf16(
            false, aq[kd].v, false, bk.v, (short)0, sc[ni].v, false, false);
      }
    }
    // ---- scale + causal mask ----
    #pragma unroll
    for (int ni = 0; ni < 2; ++ni) {
      const int col = kb + ni * 16 + l16;
      #pragma unroll
      for (int r = 0; r < 8; ++r) {
        const int rowg = q0 + r + 8 * half;
        const float sv = sc[ni].f[r] * scale;
        sc[ni].f[r] = (col <= rowg) ? sv : -1e30f;
      }
    }
    // ---- online softmax update (row = r + 8*half lives across 16 lanes) ----
    #pragma unroll
    for (int r = 0; r < 8; ++r) {
      float m = fmaxf(sc[0].f[r], sc[1].f[r]);
      m = fmaxf(m, __shfl_xor(m, 8));
      m = fmaxf(m, __shfl_xor(m, 4));
      m = fmaxf(m, __shfl_xor(m, 2));
      m = fmaxf(m, __shfl_xor(m, 1));
      const float nm  = fmaxf(rmax[r], m);
      const float cor = __expf(rmax[r] - nm);
      const float p0 = __expf(sc[0].f[r] - nm);
      const float p1 = __expf(sc[1].f[r] - nm);
      sc[0].f[r] = p0; sc[1].f[r] = p1;
      float ps = p0 + p1;
      ps += __shfl_xor(ps, 8);
      ps += __shfl_xor(ps, 4);
      ps += __shfl_xor(ps, 2);
      ps += __shfl_xor(ps, 1);
      rsum[r] = rsum[r] * cor + ps;
      rmax[r] = nm;
      #pragma unroll
      for (int d4 = 0; d4 < 4; ++d4) oacc[d4].f[r] *= cor;
    }
    // ---- transpose P (C layout) -> A fragment via wave-private LDS ----
    #pragma unroll
    for (int ni = 0; ni < 2; ++ni)
      #pragma unroll
      for (int r = 0; r < 8; ++r)
        pbuf[wave][r + 8 * half][ni * 16 + l16] = (bf16)sc[ni].f[r];
    asm volatile("s_wait_dscnt 0x0" ::: "memory");   // same-wave DS RAW fence
    FragAB pf;
    {
      const bf16* pr = &pbuf[wave][l16][0];
      pf.q[0] = *(const uint4*)(pr + lo);
      pf.q[1] = *(const uint4*)(pr + lo + 16);
    }
    // ---- O[16x64] += P[16x32] @ V[32x64] ----
    #pragma unroll
    for (int d4 = 0; d4 < 4; ++d4) {
      const bf16* vrow = vp + (size_t)(d4 * 16 + l16) * kS + kb + lo;
      FragAB bv;
      bv.q[0] = *(const uint4*)(vrow);
      bv.q[1] = *(const uint4*)(vrow + 16);
      oacc[d4].v = __builtin_amdgcn_wmma_f32_16x16x32_bf16(
          false, pf.v, false, bv.v, (short)0, oacc[d4].v, false, false);
    }
  }

  // ---- finalize: O /= rowsum; store to [b*S + q][h*64 + d] ----
  #pragma unroll
  for (int d4 = 0; d4 < 4; ++d4)
    #pragma unroll
    for (int r = 0; r < 8; ++r) {
      const int rowg = q0 + r + 8 * half;
      const int colE = h * kD + d4 * 16 + l16;
      const float ov = oacc[d4].f[r] / rsum[r];
      aout[(size_t)(b * kS + rowg) * kE + colE] = (bf16)ov;
    }
}

// =====================================================================
// host-side launch
// =====================================================================
extern "C" void kernel_launch(void* const* d_in, const int* in_sizes, int n_in,
                              void* d_out, int out_size, void* d_ws, size_t ws_size,
                              hipStream_t stream) {
  const float* x     = (const float*)d_in[0];
  const float* gamma = (const float*)d_in[1];
  const float* beta  = (const float*)d_in[2];
  const float* wqkv  = (const float*)d_in[3];   // [1024][3072]
  const float* wout  = (const float*)d_in[4];   // [1024][1024]
  const float* qks   = (const float*)d_in[5];
  float* out = (float*)d_out;

  char* ws = (char*)d_ws;
  size_t off = 0;
  bf16*  h      = (bf16*)(ws + off); off += (size_t)4096 * 1024 * 2;        //  8 MB
  bf16*  wqkvT  = (bf16*)(ws + off); off += (size_t)3072 * 1024 * 2;        //  6 MB
  bf16*  woutT  = (bf16*)(ws + off); off += (size_t)1024 * 1024 * 2;        //  2 MB
  float* qkvf   = (float*)(ws + off); off += (size_t)4096 * 3072 * 4;       // 48 MB
  bf16*  qn     = (bf16*)(ws + off); off += (size_t)kB * kH * kS * kD * 2;  //  8 MB
  bf16*  kn     = (bf16*)(ws + off); off += (size_t)kB * kH * kS * kD * 2;  //  8 MB
  bf16*  vT     = (bf16*)(ws + off); off += (size_t)kB * kH * kD * kS * 2;  //  8 MB
  bf16*  aout   = (bf16*)(ws + off); off += (size_t)4096 * 1024 * 2;        //  8 MB

  // 1) LayerNorm -> bf16
  ln_kernel<<<kB * kS, 256, 0, stream>>>(x, gamma, beta, h);

  // 2) weights -> N-major bf16
  transpose_cvt_kernel<<<dim3(3072 / 32, 1024 / 32), dim3(32, 8), 0, stream>>>(wqkv, wqkvT, 1024, 3072);
  transpose_cvt_kernel<<<dim3(1024 / 32, 1024 / 32), dim3(32, 8), 0, stream>>>(wout, woutT, 1024, 1024);

  // 3) QKV GEMM: [4096,1024] x [1024,3072] -> f32
  gemm_bf16_kernel<<<dim3(3072 / 256, 4096 / 64), 256, 0, stream>>>(h, wqkvT, qkvf, 4096, 3072, 1024);

  // 4) RoPE + split + l2norm -> qn/kn/vT
  rope_split_kernel<<<kB * kS, 256, 0, stream>>>(qkvf, qn, kn, vT);

  // 5) causal flash attention
  attn_kernel<<<dim3(kS / 128, kB * kH), 256, 0, stream>>>(qn, kn, vT, qks, aout);

  // 6) output projection: [4096,1024] x [1024,1024] -> f32 d_out
  gemm_bf16_kernel<<<dim3(1024 / 256, 4096 / 64), 256, 0, stream>>>(aout, woutT, out, 4096, 1024, 1024);
}
